// DecoderBlock_45114336477867
// MI455X (gfx1250) — matrix operand order repaired and checked by reading, not compiled, not run on hardware
//
#include <hip/hip_runtime.h>
#include <hip/hip_bf16.h>

typedef __attribute__((ext_vector_type(16))) __bf16 v16bf;
typedef __attribute__((ext_vector_type(8)))  __bf16 v8bf;
typedef __attribute__((ext_vector_type(8)))  float  v8f;

#define IN_EPS 1e-6f

// ---------------------------------------------------------------------------
// Sparse conv: async-stage weight tile to LDS -> gather bf16 rows ->
// bf16 WMMA (f32 accum) -> native f32 atomic scatter.
// act : [N_in, CIN] bf16 activations
// Wt  : [K, 64, CIN] bf16 weights, N-major (pre-transposed)
// kin/kout : [K, M] index pairs, out : [N_out, 64] f32 (pre-zeroed)
// Block = 256 threads = 8 wave32; each wave computes a 16-row x 64-col tile.
// ---------------------------------------------------------------------------
template<int CIN>
__global__ __launch_bounds__(256)
void spconv_kernel(const __bf16* __restrict__ act,
                   const __bf16* __restrict__ Wt,
                   const int*    __restrict__ kin,
                   const int*    __restrict__ kout,
                   float*        __restrict__ out,
                   int M)
{
    constexpr int LDW = CIN + 8;          // padded row stride (bf16 units, 16B granular)
    __shared__ __bf16 wtile[64 * LDW];    // weight tile for this block's offset k

    const int k    = blockIdx.y;
    const int wave = threadIdx.x >> 5;    // wave32 on gfx1250
    const int lane = threadIdx.x & 31;
    const int hi   = lane >> 4;           // lane half: 0 => lanes 0-15, 1 => 16-31
    const int ln   = lane & 15;

    const int mbase = blockIdx.x * 128 + wave * 16;
    const __bf16* wbase = Wt + (size_t)k * 64 * CIN;

    // ---- async-stage W[k] (N-major, 64 x CIN) into LDS, 16B granules ----
    {
        constexpr int GPR = CIN / 8;                  // 16B granules per row
        for (int g = threadIdx.x; g < 64 * GPR; g += 256) {
            const int n = g / GPR, gi = g % GPR;
            unsigned dst = (unsigned)(size_t)&wtile[n * LDW + gi * 8];
            const __bf16* src = wbase + (size_t)n * CIN + gi * 8;
            asm volatile("global_load_async_to_lds_b128 %0, %1, off"
                         :: "v"(dst), "v"(src) : "memory");
        }
        asm volatile("s_wait_asynccnt 0x0" ::: "memory");
    }

    // Prefetch upcoming scatter indices (global_prefetch_b8 path)
    __builtin_prefetch(kout + (size_t)k * M + mbase, 0, 1);

    // A gather source: this lane supplies row M=ln of the 16x32 A tiles
    const int in_row = kin[(size_t)k * M + mbase + ln];
    const __bf16* rowp = act + (size_t)in_row * CIN;

    // Scatter rows this lane will write: C/D layout -> M = r + 8*hi
    int orow[8];
    const int ob = mbase + hi * 8;
#pragma unroll
    for (int r = 0; r < 8; ++r)
        orow[r] = kout[(size_t)k * M + ob + r];

    __syncthreads();   // weight tile visible to all waves

    v8f c[4] = {};     // 4 column tiles of 16, f32 accumulators

#pragma unroll
    for (int kc = 0; kc < CIN / 32; ++kc) {
        // A fragment (16x32 bf16): lane holds row ln,
        // K = kc*32 + hi*8 + {0..7} and + {16..23}  (ISA 16-bit A layout)
        const __bf16* ap = rowp + kc * 32 + hi * 8;
        v8bf alo = *(const v8bf*)(ap);
        v8bf ahi = *(const v8bf*)(ap + 16);
        v16bf a;
#pragma unroll
        for (int i = 0; i < 8; ++i) { a[i] = alo[i]; a[8 + i] = ahi[i]; }

#pragma unroll
        for (int nt = 0; nt < 4; ++nt) {
            // B fragment (32x16 bf16) from LDS: lane holds col n = nt*16+ln,
            // K = kc*32 + hi*8 + {0..7} and + {16..23}  (matches A layout)
            const __bf16* bp = &wtile[(nt * 16 + ln) * LDW + kc * 32 + hi * 8];
            v8bf blo = *(const v8bf*)(bp);
            v8bf bhi = *(const v8bf*)(bp + 16);
            v16bf b;
#pragma unroll
            for (int i = 0; i < 8; ++i) { b[i] = blo[i]; b[8 + i] = bhi[i]; }
            c[nt] = __builtin_amdgcn_wmma_f32_16x16x32_bf16(
                        false, a, false, b, (short)0, c[nt], false, false);
        }
    }

    // Scatter-accumulate: element (M=r+8*hi, N=nt*16+ln) lives in c[nt][r]
#pragma unroll
    for (int nt = 0; nt < 4; ++nt) {
#pragma unroll
        for (int r = 0; r < 8; ++r) {
            float* dst = out + (size_t)orow[r] * 64 + nt * 16 + ln;
            __hip_atomic_fetch_add(dst, c[nt][r],
                                   __ATOMIC_RELAXED, __HIP_MEMORY_SCOPE_AGENT);
        }
    }
}

// ---------------------------------------------------------------------------
// Per-channel sum / sumsq reduction for instance norm (64 channels)
// ---------------------------------------------------------------------------
__global__ __launch_bounds__(256)
void stats_kernel(const float* __restrict__ x, float* __restrict__ stats, int nrows)
{
    const int c = threadIdx.x & 63;
    int r = blockIdx.x * (blockDim.x >> 6) + (threadIdx.x >> 6);
    const int rstride = gridDim.x * (blockDim.x >> 6);
    float s = 0.f, s2 = 0.f;
    for (; r < nrows; r += rstride) {
        float v = x[(size_t)r * 64 + c];
        s += v; s2 += v * v;
    }
    __hip_atomic_fetch_add(&stats[c],      s,  __ATOMIC_RELAXED, __HIP_MEMORY_SCOPE_AGENT);
    __hip_atomic_fetch_add(&stats[64 + c], s2, __ATOMIC_RELAXED, __HIP_MEMORY_SCOPE_AGENT);
}

// ---------------------------------------------------------------------------
// Fused instance-norm + ELU (+ optional f32 skip add), f32 -> bf16
// ---------------------------------------------------------------------------
__global__ __launch_bounds__(256)
void normelu_bf16_kernel(const float* __restrict__ x, const float* __restrict__ stats,
                         const float* __restrict__ enc, __bf16* __restrict__ out, int nrows)
{
    size_t i = (size_t)blockIdx.x * blockDim.x + threadIdx.x;
    if (i >= (size_t)nrows * 64) return;
    const int c = (int)(i & 63);
    const float inv = 1.f / (float)nrows;
    const float mu  = stats[c] * inv;
    const float var = stats[64 + c] * inv - mu * mu;
    const float rs  = rsqrtf(var + IN_EPS);
    float v = (x[i] - mu) * rs;
    v = v > 0.f ? v : (expf(v) - 1.f);          // ELU(alpha=1)
    if (enc) v += enc[i];                        // skip connection
    out[i] = (__bf16)v;
}

// Same, but keep f32 (final layer feeding the pruning head / d_out)
__global__ __launch_bounds__(256)
void normelu_f32_kernel(const float* __restrict__ x, const float* __restrict__ stats,
                        float* __restrict__ out, int nrows)
{
    size_t i = (size_t)blockIdx.x * blockDim.x + threadIdx.x;
    if (i >= (size_t)nrows * 64) return;
    const int c = (int)(i & 63);
    const float inv = 1.f / (float)nrows;
    const float mu  = stats[c] * inv;
    const float var = stats[64 + c] * inv - mu * mu;
    const float rs  = rsqrtf(var + IN_EPS);
    float v = (x[i] - mu) * rs;
    out[i] = v > 0.f ? v : (expf(v) - 1.f);
}

// ---------------------------------------------------------------------------
// Pruning head: keep = (y @ Wp + bp) > 0 ; y *= keep. One wave32 per row,
// butterfly reduction via __shfl_xor (wave32-correct).
// ---------------------------------------------------------------------------
__global__ __launch_bounds__(256)
void prune_kernel(float* __restrict__ y, const float* __restrict__ Wp,
                  const float* __restrict__ bp, float* __restrict__ keep, int nrows)
{
    const int lane = threadIdx.x & 31;
    int row = blockIdx.x * (blockDim.x >> 5) + (threadIdx.x >> 5);
    const int rstride = gridDim.x * (blockDim.x >> 5);
    const float w0 = Wp[lane], w1 = Wp[32 + lane], b = bp[0];
    for (; row < nrows; row += rstride) {
        float v0 = y[(size_t)row * 64 + lane];
        float v1 = y[(size_t)row * 64 + 32 + lane];
        float d  = v0 * w0 + v1 * w1;
#pragma unroll
        for (int off = 16; off > 0; off >>= 1) d += __shfl_xor(d, off, 32);
        const float kf = (d + b) > 0.f ? 1.f : 0.f;
        y[(size_t)row * 64 + lane]      = v0 * kf;
        y[(size_t)row * 64 + 32 + lane] = v1 * kf;
        if (lane == 0) keep[row] = kf;
    }
}

// ---------------------------------------------------------------------------
// Conversions: f32 activations -> bf16, and W[k][c][n] -> Wt[k][n][c] bf16
// ---------------------------------------------------------------------------
__global__ __launch_bounds__(256)
void f32_to_bf16_kernel(const float* __restrict__ in, __bf16* __restrict__ out, size_t n)
{
    size_t i = (size_t)blockIdx.x * blockDim.x + threadIdx.x;
    if (i < n) out[i] = (__bf16)in[i];
}

__global__ __launch_bounds__(256)
void wtrans_kernel(const float* __restrict__ W, __bf16* __restrict__ Wt, int K, int CIN)
{
    size_t i = (size_t)blockIdx.x * blockDim.x + threadIdx.x;
    size_t total = (size_t)K * CIN * 64;
    if (i >= total) return;
    int k   = (int)(i / ((size_t)CIN * 64));
    int rem = (int)(i % ((size_t)CIN * 64));
    int c = rem / 64, n = rem % 64;
    Wt[((size_t)k * 64 + n) * CIN + c] = (__bf16)W[i];
}

// ---------------------------------------------------------------------------
extern "C" void kernel_launch(void* const* d_in, const int* in_sizes, int n_in,
                              void* d_out, int out_size, void* d_ws, size_t ws_size,
                              hipStream_t stream)
{
    const float* input_feats = (const float*)d_in[0];
    const float* encoder     = (const float*)d_in[1];
    const float* W1          = (const float*)d_in[2];
    const float* W2          = (const float*)d_in[3];
    const float* W3          = (const float*)d_in[4];
    const float* Wp          = (const float*)d_in[5];
    const float* bp          = (const float*)d_in[6];
    const int* kin1  = (const int*)d_in[7];
    const int* kout1 = (const int*)d_in[8];
    const int* kin2  = (const int*)d_in[9];
    const int* kout2 = (const int*)d_in[10];
    const int* kin3  = (const int*)d_in[11];
    const int* kout3 = (const int*)d_in[12];

    constexpr int N_IN = 65536, N_UP = 262144, N_OUT = 131072;
    constexpr int K1 = 27, K2 = 27, K3 = 16;
    constexpr int M1 = 65536, M2 = 131072, M3 = 131072;

    // Workspace carve-out (256B aligned)
    char* ws = (char*)d_ws;
    size_t off = 0;
    auto carve = [&](size_t bytes) -> void* {
        void* p = ws + off;
        off = (off + bytes + 255) & ~(size_t)255;
        return p;
    };
    float*  bufA  = (float*)carve((size_t)N_UP * 64 * sizeof(float));   // shared f32 accumulator
    __bf16* act   = (__bf16*)carve((size_t)N_UP * 64 * sizeof(__bf16)); // bf16 activations
    __bf16* inb   = (__bf16*)carve((size_t)N_IN * 128 * sizeof(__bf16));
    __bf16* w1t   = (__bf16*)carve((size_t)K1 * 64 * 128 * sizeof(__bf16));
    __bf16* w2t   = (__bf16*)carve((size_t)K2 * 64 * 64 * sizeof(__bf16));
    __bf16* w3t   = (__bf16*)carve((size_t)K3 * 64 * 64 * sizeof(__bf16));
    float*  stats = (float*)carve(3 * 128 * sizeof(float));

    float* y    = (float*)d_out;          // [N_OUT*64]
    float* keep = y + (size_t)N_OUT * 64; // [N_OUT] as 0/1 floats

    hipMemsetAsync(stats, 0, 3 * 128 * sizeof(float), stream);

    // Precision / layout conversions
    { size_t n = (size_t)N_IN * 128;
      f32_to_bf16_kernel<<<(unsigned)((n + 255) / 256), 256, 0, stream>>>(input_feats, inb, n); }
    { size_t n = (size_t)K1 * 128 * 64;
      wtrans_kernel<<<(unsigned)((n + 255) / 256), 256, 0, stream>>>(W1, w1t, K1, 128); }
    { size_t n = (size_t)K2 * 64 * 64;
      wtrans_kernel<<<(unsigned)((n + 255) / 256), 256, 0, stream>>>(W2, w2t, K2, 64); }
    { size_t n = (size_t)K3 * 64 * 64;
      wtrans_kernel<<<(unsigned)((n + 255) / 256), 256, 0, stream>>>(W3, w3t, K3, 64); }

    // ---- conv-transpose 1: [65536,128] -> [262144,64] ----
    hipMemsetAsync(bufA, 0, (size_t)N_UP * 64 * sizeof(float), stream);
    spconv_kernel<128><<<dim3(M1 / 128, K1), 256, 0, stream>>>(inb, w1t, kin1, kout1, bufA, M1);
    stats_kernel<<<512, 256, 0, stream>>>(bufA, stats, N_UP);
    normelu_bf16_kernel<<<(unsigned)(((size_t)N_UP * 64 + 255) / 256), 256, 0, stream>>>(
        bufA, stats, nullptr, act, N_UP);

    // ---- conv 2: [262144,64] -> [262144,64] ; then IN+ELU + skip(encoder) ----
    hipMemsetAsync(bufA, 0, (size_t)N_UP * 64 * sizeof(float), stream);
    spconv_kernel<64><<<dim3(M2 / 128, K2), 256, 0, stream>>>(act, w2t, kin2, kout2, bufA, M2);
    stats_kernel<<<512, 256, 0, stream>>>(bufA, stats + 128, N_UP);
    normelu_bf16_kernel<<<(unsigned)(((size_t)N_UP * 64 + 255) / 256), 256, 0, stream>>>(
        bufA, stats + 128, encoder, act, N_UP);

    // ---- conv 3: [262144,64] -> [131072,64] ; IN+ELU -> d_out ----
    hipMemsetAsync(bufA, 0, (size_t)N_OUT * 64 * sizeof(float), stream);
    spconv_kernel<64><<<dim3(M3 / 128, K3), 256, 0, stream>>>(act, w3t, kin3, kout3, bufA, M3);
    stats_kernel<<<512, 256, 0, stream>>>(bufA, stats + 256, N_OUT);
    normelu_f32_kernel<<<(unsigned)(((size_t)N_OUT * 64 + 255) / 256), 256, 0, stream>>>(
        bufA, stats + 256, y, N_OUT);

    // ---- pruning head ----
    prune_kernel<<<1024, 256, 0, stream>>>(y, Wp, bp, keep, N_OUT);
}
